// Mamba_36601711296957
// MI455X (gfx1250) — compile-verified, matrix operand order checked by reading, not run on hardware
//
#include <hip/hip_runtime.h>
#include <hip/hip_bf16.h>
#include <math.h>

// ---------------- model dims ----------------
#define D_MODEL 1024
#define N_LAYER 2
#define D_STATE 16
#define D_CONV  4
#define D_INNER 2048
#define DT_RANK 64
#define VOCAB   32000
#define BATCH   2
#define SEQLEN  1024
#define TOKENS  (BATCH * SEQLEN)   // 2048

typedef __attribute__((ext_vector_type(16))) __bf16 v16bf;
typedef __attribute__((ext_vector_type(8)))  __bf16 v8bf;
typedef __attribute__((ext_vector_type(4)))  __bf16 v4bf;
typedef __attribute__((ext_vector_type(8)))  float  v8f;

// ==================================================================
// Big WMMA GEMM: C[M,N] = act(A[M,K] @ W[N,K]^T + bias[N]) + residual
//   - 256 threads = 8 waves, block tile 128(M) x 128(N), K step 32
//   - double-buffered LDS (32 KB): global loads for tile k+32 are
//     issued before the WMMAs of tile k, so s_wait_loadcnt lands after
//     the matrix work (HBM latency hidden behind v_wmma)
//   - wave grid 2(M) x 4(N); each wave: 4x2 tiles of 16x16
//     => 8 v_wmma_f32_16x16x32_bf16 per K step
// Requirements: M % 128 == 0 (grid.y), N % 128 == 0 (grid.x),
//               K % 32 == 0, lda/ldw % 4 == 0.
// Fragment layout per CDNA5 ISA 7.12.2 (16-bit, 16x32):
//   lane l: row = l&15, half = l>>4
//   elems 0..7  <- K = k0 + 8*half + 0..7   (16B contiguous in LDS)
//   elems 8..15 <- K = k0 + 16 + 8*half + 0..7
// C/D: elem i -> C[m0 + i + 8*half][n0 + (l&15)]
// ==================================================================
template<int ACT, bool HAS_BIAS, bool HAS_RES>
__global__ __launch_bounds__(256)
void gemm_wmma_big(const float* __restrict__ A, int lda,
                   const float* __restrict__ W, int ldw,
                   float* __restrict__ C, int ldc,
                   const float* __restrict__ residual,
                   const float* __restrict__ bias,
                   int K)
{
    __shared__ __bf16 lsA[2][128 * 32];
    __shared__ __bf16 lsB[2][128 * 32];

    const int tid  = threadIdx.x;
    const int lane = tid & 31;
    const int wave = tid >> 5;      // 0..7
    const int wm   = wave >> 2;     // 0..1  (M direction, 64 rows each)
    const int wn   = wave & 3;      // 0..3  (N direction, 32 cols each)
    const int half = lane >> 4;     // 0..1
    const int r    = lane & 15;

    const int m_blk = blockIdx.y * 128;
    const int n_blk = blockIdx.x * 128;

    // this thread's 4 slots of the cooperative 128x32 stage (8 float4/row)
    const int srow0 = tid >> 3;          // + it*32
    const int sc4   = tid & 7;

    float4 ra[4], rw[4];

    auto stage_load = [&](int k0) {
#pragma unroll
        for (int it = 0; it < 4; ++it) {
            const int row = srow0 + it * 32;
            ra[it] = *(const float4*)(A + (size_t)(m_blk + row) * lda + k0 + sc4 * 4);
            rw[it] = *(const float4*)(W + (size_t)(n_blk + row) * ldw + k0 + sc4 * 4);
        }
    };
    auto stage_store = [&](int buf) {
#pragma unroll
        for (int it = 0; it < 4; ++it) {
            const int row = srow0 + it * 32;
            v4bf pa, pb;
            pa[0] = (__bf16)ra[it].x; pa[1] = (__bf16)ra[it].y;
            pa[2] = (__bf16)ra[it].z; pa[3] = (__bf16)ra[it].w;
            pb[0] = (__bf16)rw[it].x; pb[1] = (__bf16)rw[it].y;
            pb[2] = (__bf16)rw[it].z; pb[3] = (__bf16)rw[it].w;
            *(v4bf*)&lsA[buf][row * 32 + sc4 * 4] = pa;
            *(v4bf*)&lsB[buf][row * 32 + sc4 * 4] = pb;
        }
    };

    v8f acc[4][2] = {};

    // prologue: stage tile k=0 into buffer 0
    stage_load(0);
    stage_store(0);
    __syncthreads();

    for (int k0 = 0; k0 < K; k0 += 32) {
        const int  buf      = (k0 >> 5) & 1;
        const bool has_next = (k0 + 32 < K);

        // issue next tile's global loads BEFORE consuming this tile
        if (has_next) stage_load(k0 + 32);

        // ---- fragment loads from LDS (ds_load_b128 x2 per fragment) ----
        v16bf fa[4], fb[2];
#pragma unroll
        for (int mt = 0; mt < 4; ++mt) {
            const int row = wm * 64 + mt * 16 + r;
            v8bf lo = *(const v8bf*)&lsA[buf][row * 32 + half * 8];
            v8bf hi = *(const v8bf*)&lsA[buf][row * 32 + 16 + half * 8];
            fa[mt] = __builtin_shufflevector(lo, hi, 0,1,2,3,4,5,6,7,8,9,10,11,12,13,14,15);
        }
#pragma unroll
        for (int nt = 0; nt < 2; ++nt) {
            const int row = wn * 32 + nt * 16 + r;
            v8bf lo = *(const v8bf*)&lsB[buf][row * 32 + half * 8];
            v8bf hi = *(const v8bf*)&lsB[buf][row * 32 + 16 + half * 8];
            fb[nt] = __builtin_shufflevector(lo, hi, 0,1,2,3,4,5,6,7,8,9,10,11,12,13,14,15);
        }

        // ---- 8 WMMAs (latency of next-tile global loads hides here) ----
#pragma unroll
        for (int mt = 0; mt < 4; ++mt)
#pragma unroll
            for (int nt = 0; nt < 2; ++nt)
                acc[mt][nt] = __builtin_amdgcn_wmma_f32_16x16x32_bf16(
                    false, fa[mt], false, fb[nt], (short)0, acc[mt][nt], false, false);

        // convert + park next tile in the other LDS buffer
        if (has_next) stage_store(buf ^ 1);
        __syncthreads();
    }

    // ---- epilogue ----
#pragma unroll
    for (int mt = 0; mt < 4; ++mt) {
#pragma unroll
        for (int nt = 0; nt < 2; ++nt) {
            const int n = n_blk + wn * 32 + nt * 16 + r;
            float bn = 0.0f;
            if constexpr (HAS_BIAS) bn = bias[n];
#pragma unroll
            for (int i = 0; i < 8; ++i) {
                const int m = m_blk + wm * 64 + mt * 16 + i + half * 8;
                float v = acc[mt][nt][i];
                if constexpr (HAS_BIAS) v += bn;
                if constexpr (ACT == 1)  v = (v > 20.0f) ? v : log1pf(expf(v)); // softplus
                if constexpr (HAS_RES)   v += residual[(size_t)m * ldc + n];
                C[(size_t)m * ldc + n] = v;
            }
        }
    }
}

// ==================================================================
// Small WMMA GEMM (plain): one wave per 16x16 tile. Used for N=96.
// ==================================================================
__global__ __launch_bounds__(32)
void gemm_wmma_small(const float* __restrict__ A, int lda,
                     const float* __restrict__ W, int ldw,
                     float* __restrict__ C, int ldc, int K)
{
    const int lane = threadIdx.x;
    const int half = lane >> 4;
    const int r    = lane & 15;
    const int m0   = blockIdx.y * 16;
    const int n0   = blockIdx.x * 16;

    v8f acc = {};
    for (int k0 = 0; k0 < K; k0 += 32) {
        const float* arow = A + (size_t)(m0 + r) * lda + k0 + half * 8;
        const float* wrow = W + (size_t)(n0 + r) * ldw + k0 + half * 8;
        v16bf af, bfrag;
#pragma unroll
        for (int e = 0; e < 8; ++e) {
            af[e]        = (__bf16)arow[e];
            af[e + 8]    = (__bf16)arow[16 + e];
            bfrag[e]     = (__bf16)wrow[e];
            bfrag[e + 8] = (__bf16)wrow[16 + e];
        }
        acc = __builtin_amdgcn_wmma_f32_16x16x32_bf16(
            false, af, false, bfrag, (short)0, acc, false, false);
    }
    const int n = n0 + r;
#pragma unroll
    for (int i = 0; i < 8; ++i) {
        const int m = m0 + i + half * 8;
        C[(size_t)m * ldc + n] = acc[i];
    }
}

// ---------------- embedding gather ----------------
__global__ void embed_kernel(const int* __restrict__ x,
                             const float* __restrict__ emb,
                             float* __restrict__ h)
{
    size_t idx = (size_t)blockIdx.x * blockDim.x + threadIdx.x;
    if (idx >= (size_t)TOKENS * D_MODEL) return;
    size_t t = idx / D_MODEL;
    int    d = (int)(idx % D_MODEL);
    h[idx] = emb[(size_t)x[t] * D_MODEL + d];
}

// ---------------- RMSNorm (one block per token) ----------------
__global__ __launch_bounds__(256)
void rmsnorm_kernel(const float* __restrict__ h,
                    const float* __restrict__ w,
                    float* __restrict__ out)
{
    const int t = blockIdx.x;
    const float* row = h + (size_t)t * D_MODEL;
    __shared__ float red[8];
    __shared__ float s_scale;

    float s = 0.0f;
    for (int i = threadIdx.x; i < D_MODEL; i += 256) {
        float v = row[i];
        s += v * v;
    }
#pragma unroll
    for (int off = 16; off > 0; off >>= 1)
        s += __shfl_down(s, off, 32);
    if ((threadIdx.x & 31) == 0) red[threadIdx.x >> 5] = s;
    __syncthreads();
    if (threadIdx.x == 0) {
        float tot = 0.0f;
#pragma unroll
        for (int i = 0; i < 8; ++i) tot += red[i];
        s_scale = rsqrtf(tot / (float)D_MODEL + 1e-5f);
    }
    __syncthreads();
    const float sc = s_scale;
    for (int i = threadIdx.x; i < D_MODEL; i += 256)
        out[(size_t)t * D_MODEL + i] = row[i] * sc * w[i];
}

// ---------------- causal depthwise conv (width 4) + SiLU ----------------
__global__ void conv_silu_kernel(const float* __restrict__ xz,   // (T, 2*D_INNER), xi = cols [0,2048)
                                 const float* __restrict__ cw,   // (D_INNER, 4)
                                 const float* __restrict__ cb,   // (D_INNER)
                                 float* __restrict__ xc)         // (T, D_INNER)
{
    size_t idx = (size_t)blockIdx.x * blockDim.x + threadIdx.x;
    if (idx >= (size_t)TOKENS * D_INNER) return;
    int    d = (int)(idx % D_INNER);
    size_t t = idx / D_INNER;          // b*SEQLEN + l
    int    l = (int)(t % SEQLEN);

    float acc = cb[d];
#pragma unroll
    for (int j = 0; j < D_CONV; ++j) {
        int ls = l - (D_CONV - 1) + j;
        if (ls >= 0)
            acc += xz[(t - (size_t)l + ls) * (2 * D_INNER) + d] * cw[d * D_CONV + j];
    }
    xc[idx] = acc / (1.0f + expf(-acc));   // silu
}

// ---------------- selective scan: one thread per (b, d) ----------------
__global__ void scan_kernel(const float* __restrict__ dt,    // (T, D_INNER)
                            const float* __restrict__ xc,    // (T, D_INNER)
                            const float* __restrict__ dbl,   // (T, 96): [0,64)=dt_low [64,80)=B [80,96)=C
                            const float* __restrict__ xz,    // (T, 2*D_INNER), z = cols [2048,4096)
                            const float* __restrict__ A_log, // (D_INNER, D_STATE)
                            const float* __restrict__ Dp,    // (D_INNER)
                            float* __restrict__ y)           // (T, D_INNER)
{
    int idx = blockIdx.x * blockDim.x + threadIdx.x;
    if (idx >= BATCH * D_INNER) return;
    const int b = idx / D_INNER;
    const int d = idx % D_INNER;

    float Av[D_STATE];
#pragma unroll
    for (int n = 0; n < D_STATE; ++n)
        Av[n] = -expf(A_log[(size_t)d * D_STATE + n]);
    const float Dd = Dp[d];

    float state[D_STATE];
#pragma unroll
    for (int n = 0; n < D_STATE; ++n) state[n] = 0.0f;

    for (int l = 0; l < SEQLEN; ++l) {
        const size_t t   = (size_t)b * SEQLEN + l;
        const float  dtv = dt[t * D_INNER + d];
        const float  xv  = xc[t * D_INNER + d];
        const float* Brow = dbl + t * 96 + DT_RANK;
        const float* Crow = dbl + t * 96 + DT_RANK + D_STATE;

        float acc = 0.0f;
#pragma unroll
        for (int n = 0; n < D_STATE; ++n) {
            float a = expf(dtv * Av[n]);
            state[n] = a * state[n] + dtv * Brow[n] * xv;
            acc += state[n] * Crow[n];
        }
        acc += xv * Dd;
        const float zv = xz[t * (2 * D_INNER) + D_INNER + d];
        const float sz = zv / (1.0f + expf(-zv));
        y[t * D_INNER + d] = acc * sz;
    }
}

// ------------------------------------------------------------------
extern "C" void kernel_launch(void* const* d_in, const int* in_sizes, int n_in,
                              void* d_out, int out_size, void* d_ws, size_t ws_size,
                              hipStream_t stream)
{
    const int*   x          = (const int*)  d_in[0];
    const float* embedding  = (const float*)d_in[1];
    const float* norm_w     = (const float*)d_in[2];
    const float* in_proj_w  = (const float*)d_in[3];
    const float* conv_w     = (const float*)d_in[4];
    const float* conv_b     = (const float*)d_in[5];
    const float* x_proj_w   = (const float*)d_in[6];
    const float* dt_proj_w  = (const float*)d_in[7];
    const float* dt_bias    = (const float*)d_in[8];
    const float* A_log      = (const float*)d_in[9];
    const float* Dp         = (const float*)d_in[10];
    const float* out_proj_w = (const float*)d_in[11];
    const float* final_norm = (const float*)d_in[12];
    float*       logits     = (float*)d_out;

    // workspace layout (floats)
    float* ws  = (float*)d_ws;
    float* h   = ws;                                  // T * 1024
    float* u   = h   + (size_t)TOKENS * D_MODEL;      // T * 1024
    float* xzb = u   + (size_t)TOKENS * D_MODEL;      // T * 4096
    float* xcb = xzb + (size_t)TOKENS * 2 * D_INNER;  // T * 2048
    float* dbl = xcb + (size_t)TOKENS * D_INNER;      // T * 96
    float* dtb = dbl + (size_t)TOKENS * 96;           // T * 2048
    float* yb  = dtb + (size_t)TOKENS * D_INNER;      // T * 2048

    // 1) embedding gather
    {
        size_t n = (size_t)TOKENS * D_MODEL;
        embed_kernel<<<dim3((unsigned)((n + 255) / 256)), dim3(256), 0, stream>>>(x, embedding, h);
    }

    // 2) layers
    for (int layer = 0; layer < N_LAYER; ++layer) {
        const float* l_norm  = norm_w     + (size_t)layer * D_MODEL;
        const float* l_inprj = in_proj_w  + (size_t)layer * (2 * D_INNER) * D_MODEL;
        const float* l_cw    = conv_w     + (size_t)layer * D_INNER * D_CONV;
        const float* l_cb    = conv_b     + (size_t)layer * D_INNER;
        const float* l_xprj  = x_proj_w   + (size_t)layer * (DT_RANK + 2 * D_STATE) * D_INNER;
        const float* l_dtprj = dt_proj_w  + (size_t)layer * D_INNER * DT_RANK;
        const float* l_dtb   = dt_bias    + (size_t)layer * D_INNER;
        const float* l_Alog  = A_log      + (size_t)layer * D_INNER * D_STATE;
        const float* l_D     = Dp         + (size_t)layer * D_INNER;
        const float* l_oprj  = out_proj_w + (size_t)layer * D_MODEL * D_INNER;

        // u = rmsnorm(h)
        rmsnorm_kernel<<<dim3(TOKENS), dim3(256), 0, stream>>>(h, l_norm, u);

        // xz = u @ in_proj_w^T : (T,1024) x (4096,1024)^T -> (T,4096)
        gemm_wmma_big<0, false, false><<<dim3(2 * D_INNER / 128, TOKENS / 128), dim3(256), 0, stream>>>(
            u, D_MODEL, l_inprj, D_MODEL, xzb, 2 * D_INNER, nullptr, nullptr, D_MODEL);

        // xc = silu(depthwise_conv(xi) + b)
        {
            size_t n = (size_t)TOKENS * D_INNER;
            conv_silu_kernel<<<dim3((unsigned)((n + 255) / 256)), dim3(256), 0, stream>>>(
                xzb, l_cw, l_cb, xcb);
        }

        // dbl = xc @ x_proj_w^T : (T,2048) x (96,2048)^T -> (T,96)   [narrow N: small kernel]
        gemm_wmma_small<<<dim3((DT_RANK + 2 * D_STATE) / 16, TOKENS / 16), dim3(32), 0, stream>>>(
            xcb, D_INNER, l_xprj, D_INNER, dbl, DT_RANK + 2 * D_STATE, D_INNER);

        // dt = softplus(dt_low @ dt_proj_w^T + dt_bias) : (T,64) x (2048,64)^T -> (T,2048)
        gemm_wmma_big<1, true, false><<<dim3(D_INNER / 128, TOKENS / 128), dim3(256), 0, stream>>>(
            dbl, DT_RANK + 2 * D_STATE, l_dtprj, DT_RANK, dtb, D_INNER, nullptr, l_dtb, DT_RANK);

        // selective scan + gating -> y
        scan_kernel<<<dim3((BATCH * D_INNER + 255) / 256), dim3(256), 0, stream>>>(
            dtb, xcb, dbl, xzb, l_Alog, l_D, yb);

        // h = h + y @ out_proj_w^T : (T,2048) x (1024,2048)^T -> (T,1024), in-place residual
        gemm_wmma_big<0, false, true><<<dim3(D_MODEL / 128, TOKENS / 128), dim3(256), 0, stream>>>(
            yb, D_INNER, l_oprj, D_INNER, h, D_MODEL, h, nullptr, D_INNER);
    }

    // 3) final norm
    rmsnorm_kernel<<<dim3(TOKENS), dim3(256), 0, stream>>>(h, final_norm, u);

    // 4) logits = u @ embedding^T : (T,1024) x (32000,1024)^T -> (T,32000)
    gemm_wmma_big<0, false, false><<<dim3(VOCAB / 128, TOKENS / 128), dim3(256), 0, stream>>>(
        u, D_MODEL, embedding, D_MODEL, logits, VOCAB, nullptr, nullptr, D_MODEL);
}
